// SparseMoELayer_10642928959655
// MI455X (gfx1250) — compile-verified, matrix operand order checked by reading, not run on hardware
//
#include <hip/hip_runtime.h>
#include <hip/hip_bf16.h>

// ---------------- problem constants ----------------
#define BB 4
#define SS 2048
#define DD 1024
#define EE 8
#define FF 4096
#define TT (BB * SS)           // 8192 tokens
#define NROWS (TT * 2)         // 16384 routed rows (top-2)
#define CAP (NROWS + EE * 128) // 17408: per-expert segments padded to 128
#define CAPTILES (CAP / 128)   // 136 row tiles

typedef __bf16 bf16;
typedef bf16 v16bf __attribute__((ext_vector_type(16)));
typedef bf16 v8bf  __attribute__((ext_vector_type(8)));
typedef bf16 v4bf  __attribute__((ext_vector_type(4)));
typedef float v8f  __attribute__((ext_vector_type(8)));
typedef unsigned int u32x4 __attribute__((ext_vector_type(4)));
typedef int i32x4 __attribute__((ext_vector_type(4)));
typedef int i32x8 __attribute__((ext_vector_type(8)));

// LDS layout: K=64-deep tiles. Row = 64 bf16 (128B) + 16B pad -> 144B stride.
// 144B = 36 DW; 36*m mod 64 gives 16 distinct banks (>=4 apart): conflict-free b128.
#define TKD 64                  // K depth per LDS tile (2 WMMA k-steps)
#define LDSK 72                 // row stride in halves (64 + 8 pad)
#define TILE_BYTES (128 * LDSK * 2)  // 18432
#define LDS_AS0 0
#define LDS_AS1 (LDS_AS0 + TILE_BYTES)
#define LDS_BS0 (LDS_AS1 + TILE_BYTES)
#define LDS_BS1 (LDS_BS0 + TILE_BYTES)
#define LDS_TOK (LDS_BS1 + TILE_BYTES)   // 73728
#define LDS_WGT (LDS_TOK + 512)          // 74240
#define LDS_TOTAL (LDS_WGT + 512)        // 74752

#if __has_builtin(__builtin_amdgcn_tensor_load_to_lds)
#define USE_TDM 1
#else
#define USE_TDM 0
#endif

__device__ inline v8bf zero8() {
    v8bf z;
#pragma unroll
    for (int i = 0; i < 8; ++i) z[i] = (bf16)0.0f;
    return z;
}

// branch-free tanh-approx gelu (v_exp_f32 path, clamped so no inf/NaN special-casing)
__device__ inline float gelu_tanh(float x) {
    const float c0 = 0.7978845608028654f;  // sqrt(2/pi)
    const float c1 = 0.044715f;
    float u = c0 * (x + c1 * x * x * x);
    u = fminf(fmaxf(u, -15.0f), 15.0f);
    float t = __expf(2.0f * u);
    float th = (t - 1.0f) / (t + 1.0f);
    return 0.5f * x * (1.0f + th);
}

// ---------------- Tensor Data Mover: 2D tile -> LDS ----------------
// Loads 128 rows x 64 bf16 (128B) with global row stride `stride_elems`,
// into LDS at byte offset lds_off with a 144B padded row stride
// (pad_enable, pad_interval=32 DW written, pad_amount=4 DW).
__device__ inline void tdm_load_2d(unsigned int lds_off, const void* gaddr,
                                   unsigned int stride_elems) {
#if USE_TDM
    unsigned long long ga = (unsigned long long)(uintptr_t)gaddr;
    u32x4 g0;
    g0[0] = 1u;                                   // count=1, user descriptor, no gather
    g0[1] = lds_off;                              // lds_addr
    g0[2] = (unsigned int)(ga & 0xFFFFFFFFu);     // global_addr[31:0]
    g0[3] = (unsigned int)((ga >> 32) & 0x1FFFFFFu) | (2u << 30);  // addr[56:32] | type=2
    i32x8 g1;
    g1[0] = (1 << 16)        // data_size = 2 bytes
          | (1 << 20)        // pad_enable
          | (4 << 22)        // pad_interval: 32 DWORDs (128B) written...
          | (3 << 25);       // ...then pad 4 DWORDs (16B) -> 144B LDS stride
    g1[1] = (int)((0x40000000u & 0xFFFFu) << 16) | 0;                     // tensor_dim0 lo16 (huge)
    g1[2] = (int)((0x40000000u >> 16) | ((0x40000000u & 0xFFFFu) << 16)); // dim0 hi / dim1 lo
    g1[3] = (int)((0x40000000u >> 16) | ((unsigned)TKD << 16));           // dim1 hi | tile_dim0=64
    g1[4] = (int)128;                                                     // tile_dim1=128, tile_dim2=0
    g1[5] = (int)stride_elems;                                            // tensor_dim0_stride lo32
    g1[6] = 0;                                                            // stride hi16, dim1_stride lo16
    g1[7] = 0;
    i32x4 z4 = {0, 0, 0, 0};
#if defined(__clang_major__) && (__clang_major__ >= 23)
    i32x8 z8 = {0, 0, 0, 0, 0, 0, 0, 0};
    __builtin_amdgcn_tensor_load_to_lds(g0, g1, z4, z4, z8, 0);
#else
    __builtin_amdgcn_tensor_load_to_lds(g0, g1, z4, z4, 0);
#endif
#else
    (void)lds_off; (void)gaddr; (void)stride_elems;
#endif
}

__device__ inline void tdm_wait() {
#if USE_TDM
    __builtin_amdgcn_s_wait_tensorcnt(0);
#endif
}

// fallback cooperative tile fill (used only if TDM builtin is unavailable)
// each thread covers one 64B quarter-segment pair: row = tid>>1, 4 x 16B at (tid&1)*64B
__device__ inline void coop_fill(bf16* dst, const bf16* src_base, int lr, int ls,
                                 size_t row_stride) {
    const bf16* src = src_base + (size_t)lr * row_stride + ls;
    bf16* d = dst + lr * LDSK + ls;
#pragma unroll
    for (int q = 0; q < 4; ++q)
        *reinterpret_cast<v8bf*>(d + q * 8) = *reinterpret_cast<const v8bf*>(src + q * 8);
}

// ---------------- fp32 -> bf16 elementwise (x) ----------------
__global__ void cvt_x_kernel(const float* __restrict__ x, bf16* __restrict__ xb, int n4) {
    int i = blockIdx.x * blockDim.x + threadIdx.x;
    if (i >= n4) return;
    float4 v = reinterpret_cast<const float4*>(x)[i];
    v4bf o;
    o[0] = (bf16)v.x; o[1] = (bf16)v.y; o[2] = (bf16)v.z; o[3] = (bf16)v.w;
    reinterpret_cast<v4bf*>(xb)[i] = o;
}

// ---------------- fp32 -> bf16 tiled transpose ----------------
// in: [E][R][C] fp32  ->  out: [E][C][R] bf16
__global__ void cvt_transpose_kernel(const float* __restrict__ in, bf16* __restrict__ out,
                                     int R, int C) {
    __shared__ float tile[32][33];
    int e  = blockIdx.z;
    int c0 = blockIdx.x * 32;
    int r0 = blockIdx.y * 32;
    int tx = threadIdx.x & 31;
    int ty = threadIdx.x >> 5;  // 0..7
#pragma unroll
    for (int i = 0; i < 4; ++i) {
        int r = r0 + ty + i * 8;
        tile[ty + i * 8][tx] = in[((size_t)e * R + r) * C + (c0 + tx)];
    }
    __syncthreads();
#pragma unroll
    for (int i = 0; i < 4; ++i) {
        int c = c0 + ty + i * 8;
        out[((size_t)e * C + c) * R + (r0 + tx)] = (bf16)tile[tx][ty + i * 8];
    }
}

// ---------------- router: logits, softmax, top-2, counts ----------------
__global__ void router_kernel(const float* __restrict__ x, const float* __restrict__ Wg,
                              const float* __restrict__ bg,
                              int* __restrict__ top_i, float* __restrict__ top_w,
                              int* __restrict__ counts) {
    int wave  = threadIdx.x >> 5;
    int lane  = threadIdx.x & 31;
    int token = blockIdx.x * (blockDim.x >> 5) + wave;
    if (token >= TT) return;
    const float* xr = x + (size_t)token * DD;
    float acc[EE];
#pragma unroll
    for (int e = 0; e < EE; ++e) acc[e] = 0.0f;
    for (int d = lane; d < DD; d += 32) {
        float xv = xr[d];
        const float* wr = Wg + (size_t)d * EE;
#pragma unroll
        for (int e = 0; e < EE; ++e) acc[e] += xv * wr[e];
    }
#pragma unroll
    for (int e = 0; e < EE; ++e) {
#pragma unroll
        for (int off = 16; off > 0; off >>= 1) acc[e] += __shfl_xor(acc[e], off, 32);
    }
    if (lane == 0) {
        float lg[EE];
        float mx = -1e30f;
#pragma unroll
        for (int e = 0; e < EE; ++e) { lg[e] = acc[e] + bg[e]; mx = fmaxf(mx, lg[e]); }
        float sum = 0.0f;
#pragma unroll
        for (int e = 0; e < EE; ++e) { lg[e] = __expf(lg[e] - mx); sum += lg[e]; }
        float inv = 1.0f / sum;
#pragma unroll
        for (int e = 0; e < EE; ++e) lg[e] *= inv;  // probs
        int i1 = 0; float p1 = lg[0];
#pragma unroll
        for (int e = 1; e < EE; ++e) if (lg[e] > p1) { p1 = lg[e]; i1 = e; }
        int i2 = -1; float p2 = -1.0f;
#pragma unroll
        for (int e = 0; e < EE; ++e) if (e != i1 && lg[e] > p2) { p2 = lg[e]; i2 = e; }
        float s = 1.0f / (p1 + p2 + 1e-6f);
        top_i[token * 2 + 0] = i1;
        top_i[token * 2 + 1] = i2;
        top_w[token * 2 + 0] = p1 * s;
        top_w[token * 2 + 1] = p2 * s;
        atomicAdd(&counts[i1], 1);
        atomicAdd(&counts[i2], 1);
    }
}

// ---------------- offsets: 128-aligned per-expert segments ----------------
__global__ void offsets_kernel(const int* __restrict__ counts, int* __restrict__ offs) {
    if (threadIdx.x == 0 && blockIdx.x == 0) {
        int o = 0;
        for (int e = 0; e < EE; ++e) {
            offs[e] = o;
            o += (counts[e] + 127) & ~127;
        }
        offs[EE] = o;
    }
}

// ---------------- scatter tokens into expert row segments ----------------
__global__ void scatter_kernel(const int* __restrict__ top_i, const float* __restrict__ top_w,
                               const int* __restrict__ offs, int* __restrict__ cursors,
                               int* __restrict__ row_token, float* __restrict__ row_weight) {
    int t = blockIdx.x * blockDim.x + threadIdx.x;
    if (t >= TT) return;
#pragma unroll
    for (int k = 0; k < 2; ++k) {
        int e   = top_i[t * 2 + k];
        int pos = atomicAdd(&cursors[e], 1);
        int r   = offs[e] + pos;
        row_token[r]  = t;
        row_weight[r] = top_w[t * 2 + k];
    }
}

// ---------------- WMMA fragment helper ----------------
// 16-bit A/B fragment (16x32): lane<16 -> kb=0, lane>=16 -> kb=8.
// elements 0..7  = K kb..kb+7 (16B contiguous); 8..15 = K kb+16..kb+23 (16B contiguous)
__device__ inline v16bf load_frag(const bf16* row, int kb) {
    v8bf lo = *reinterpret_cast<const v8bf*>(row + kb);
    v8bf hi = *reinterpret_cast<const v8bf*>(row + kb + 16);
    return __builtin_shufflevector(lo, hi, 0, 1, 2, 3, 4, 5, 6, 7,
                                   8, 9, 10, 11, 12, 13, 14, 15);
}

// shared 16-WMMA compute step over one K=64 resident tile pair
template <typename AccT>
__device__ inline void compute_tile(const bf16* Ab, const bf16* Bb, int wm, int wn,
                                    int lm, int kb, AccT& acc) {
#pragma unroll
    for (int ks = 0; ks < 2; ++ks) {
        const int ko = ks * 32;
        v16bf a[2], b[4];
#pragma unroll
        for (int i = 0; i < 2; ++i)
            a[i] = load_frag(Ab + (wm * 32 + i * 16 + lm) * LDSK + ko, kb);
#pragma unroll
        for (int j = 0; j < 4; ++j)
            b[j] = load_frag(Bb + (wn * 64 + j * 16 + lm) * LDSK + ko, kb);
#pragma unroll
        for (int i = 0; i < 2; ++i)
#pragma unroll
            for (int j = 0; j < 4; ++j)
                acc[i][j] = __builtin_amdgcn_wmma_f32_16x16x32_bf16(
                    false, a[i], false, b[j], (short)0, acc[i][j], false, false);
    }
}

// ---------------- GEMM1: H = gelu(gather(x) @ W1[e] + b1[e]) ----------------
// A: gathered bf16 x rows (regular loads, double-buffered via registers)
// B: w1t [E][F][D] (N-major) tiles via Tensor Data Mover
__global__ __launch_bounds__(256)
void gemm1_kernel(const bf16* __restrict__ xb, const bf16* __restrict__ w1t,
                  const float* __restrict__ b1, const int* __restrict__ row_token,
                  const int* __restrict__ offs, bf16* __restrict__ H) {
    extern __shared__ char smem[];
    int*  s_tok = (int*)(smem + LDS_TOK);

    const int tid  = threadIdx.x;
    const int row0 = blockIdx.x * 128;
    const int f0   = blockIdx.y * 128;

    int e = 0;
#pragma unroll
    for (int i = 1; i < EE; ++i)
        if (offs[i] <= row0) e = i;

    if (tid < 128) s_tok[tid] = row_token[row0 + tid];
    __syncthreads();

    const int lane = tid & 31;
    const int wave = tid >> 5;
    const int wm   = wave >> 1;        // 0..3 -> 32-row band
    const int wn   = wave & 1;         // 0..1 -> 64-col band
    const int lm   = lane & 15;
    const int kb   = (lane < 16) ? 0 : 8;

    v8f acc[2][4];
#pragma unroll
    for (int i = 0; i < 2; ++i)
#pragma unroll
        for (int j = 0; j < 4; ++j)
#pragma unroll
            for (int v = 0; v < 8; ++v) acc[i][j][v] = 0.0f;

    const int lr = tid >> 1;          // 0..127 tile row for loading
    const int ls = (tid & 1) * 32;    // element offset 0 / 32 (64B halves of a 128B row)
    const int tk = s_tok[lr];
    const bf16* aptr = (tk >= 0) ? (xb + (size_t)tk * DD + ls) : nullptr;
    const bf16* w1e  = w1t + ((size_t)e * FF + f0) * DD;  // tile base, rows stride DD

    // ---- prologue: fill buffer 0 with k0 = 0 ----
    {
        v8bf a[4];
#pragma unroll
        for (int q = 0; q < 4; ++q) a[q] = zero8();
        if (tk >= 0) {
#pragma unroll
            for (int q = 0; q < 4; ++q)
                a[q] = *reinterpret_cast<const v8bf*>(aptr + q * 8);
        }
        bf16* dst = (bf16*)(smem + LDS_AS0) + lr * LDSK + ls;
#pragma unroll
        for (int q = 0; q < 4; ++q)
            *reinterpret_cast<v8bf*>(dst + q * 8) = a[q];
#if USE_TDM
        if (wave == 0) tdm_load_2d(LDS_BS0, w1e, DD);
#else
        coop_fill((bf16*)(smem + LDS_BS0), w1e, lr, ls, DD);
#endif
        if (wave == 0) tdm_wait();
    }
    __syncthreads();

    const int NIT = DD / TKD;  // 16
    for (int it = 0; it < NIT; ++it) {
        const int cb = it & 1;
        const int nb = cb ^ 1;
        const bool has_next = (it + 1) < NIT;
        const int knext = (it + 1) * TKD;

        // issue next-tile fills before compute (overlap with WMMA)
        v8bf pa[4];
#pragma unroll
        for (int q = 0; q < 4; ++q) pa[q] = zero8();
        if (has_next) {
            if (tk >= 0) {
#pragma unroll
                for (int q = 0; q < 4; ++q)
                    pa[q] = *reinterpret_cast<const v8bf*>(aptr + knext + q * 8);
            }
#if USE_TDM
            if (wave == 0)
                tdm_load_2d(nb ? LDS_BS1 : LDS_BS0, w1e + knext, DD);
#endif
        }

        // compute from current buffer: 16 WMMAs
        compute_tile((const bf16*)(smem + (cb ? LDS_AS1 : LDS_AS0)),
                     (const bf16*)(smem + (cb ? LDS_BS1 : LDS_BS0)),
                     wm, wn, lm, kb, acc);

        if (has_next) {
            bf16* dst = (bf16*)(smem + (nb ? LDS_AS1 : LDS_AS0)) + lr * LDSK + ls;
#pragma unroll
            for (int q = 0; q < 4; ++q)
                *reinterpret_cast<v8bf*>(dst + q * 8) = pa[q];
#if !USE_TDM
            coop_fill((bf16*)(smem + (nb ? LDS_BS1 : LDS_BS0)), w1e + knext, lr, ls, DD);
#endif
        }
        if (wave == 0) tdm_wait();
        __syncthreads();
    }

    // epilogue: + b1, gelu, -> bf16 H
    const int rbase = (lane < 16) ? 0 : 8;
#pragma unroll
    for (int i = 0; i < 2; ++i) {
        int gr0 = row0 + wm * 32 + i * 16 + rbase;
#pragma unroll
        for (int j = 0; j < 4; ++j) {
            int gf = f0 + wn * 64 + j * 16 + lm;
            float bias = b1[(size_t)e * FF + gf];
#pragma unroll
            for (int v = 0; v < 8; ++v) {
                float c = acc[i][j][v] + bias;
                H[(size_t)(gr0 + v) * FF + gf] = (bf16)gelu_tanh(c);
            }
        }
    }
}

// ---------------- GEMM2: out[token] += (H @ W2[e] + b2[e]) * w ----------------
// A: H [CAP x F] and B: w2t [E][D][F] tiles both staged via Tensor Data Mover
__global__ __launch_bounds__(256)
void gemm2_kernel(const bf16* __restrict__ H, const bf16* __restrict__ w2t,
                  const float* __restrict__ b2, const int* __restrict__ row_token,
                  const float* __restrict__ row_weight, const int* __restrict__ offs,
                  float* __restrict__ out) {
    extern __shared__ char smem[];
    int*   s_tok = (int*)(smem + LDS_TOK);
    float* s_w   = (float*)(smem + LDS_WGT);

    const int tid  = threadIdx.x;
    const int row0 = blockIdx.x * 128;
    const int d0   = blockIdx.y * 128;

    int e = 0;
#pragma unroll
    for (int i = 1; i < EE; ++i)
        if (offs[i] <= row0) e = i;

    if (tid < 128) {
        s_tok[tid] = row_token[row0 + tid];
        s_w[tid]   = row_weight[row0 + tid];
    }
    __syncthreads();

    const int lane = tid & 31;
    const int wave = tid >> 5;
    const int wm   = wave >> 1;
    const int wn   = wave & 1;
    const int lm   = lane & 15;
    const int kb   = (lane < 16) ? 0 : 8;

    v8f acc[2][4];
#pragma unroll
    for (int i = 0; i < 2; ++i)
#pragma unroll
        for (int j = 0; j < 4; ++j)
#pragma unroll
            for (int v = 0; v < 8; ++v) acc[i][j][v] = 0.0f;

    const int lr = tid >> 1;
    const int ls = (tid & 1) * 32;
    const bf16* hbase = H + (size_t)row0 * FF;                // rows stride FF
    const bf16* w2e   = w2t + ((size_t)e * DD + d0) * FF;     // rows stride FF

    // ---- prologue: fill buffer 0 ----
#if USE_TDM
    if (wave == 0) {
        tdm_load_2d(LDS_AS0, hbase, FF);
        tdm_load_2d(LDS_BS0, w2e, FF);
        tdm_wait();
    }
#else
    coop_fill((bf16*)(smem + LDS_AS0), hbase, lr, ls, FF);
    coop_fill((bf16*)(smem + LDS_BS0), w2e, lr, ls, FF);
#endif
    __syncthreads();

    const int NIT = FF / TKD;  // 64
    for (int it = 0; it < NIT; ++it) {
        const int cb = it & 1;
        const int nb = cb ^ 1;
        const bool has_next = (it + 1) < NIT;
        const int knext = (it + 1) * TKD;

#if USE_TDM
        if (has_next && wave == 0) {
            tdm_load_2d(nb ? LDS_AS1 : LDS_AS0, hbase + knext, FF);
            tdm_load_2d(nb ? LDS_BS1 : LDS_BS0, w2e + knext, FF);
        }
#endif

        compute_tile((const bf16*)(smem + (cb ? LDS_AS1 : LDS_AS0)),
                     (const bf16*)(smem + (cb ? LDS_BS1 : LDS_BS0)),
                     wm, wn, lm, kb, acc);

#if !USE_TDM
        if (has_next) {
            coop_fill((bf16*)(smem + (nb ? LDS_AS1 : LDS_AS0)), hbase + knext, lr, ls, FF);
            coop_fill((bf16*)(smem + (nb ? LDS_BS1 : LDS_BS0)), w2e + knext, lr, ls, FF);
        }
#else
        if (wave == 0) tdm_wait();
#endif
        __syncthreads();
    }

    // epilogue: + b2, scale by route weight, atomic scatter-add
    // (exactly 2 float adds per out element -> bitwise deterministic)
    const int rbase = (lane < 16) ? 0 : 8;
#pragma unroll
    for (int i = 0; i < 2; ++i) {
        int lrow0 = wm * 32 + i * 16 + rbase;
#pragma unroll
        for (int j = 0; j < 4; ++j) {
            int gn = d0 + wn * 64 + j * 16 + lm;
            float bias = b2[(size_t)e * DD + gn];
#pragma unroll
            for (int v = 0; v < 8; ++v) {
                int lrow = lrow0 + v;
                int tok  = s_tok[lrow];
                if (tok >= 0) {
                    float val = (acc[i][j][v] + bias) * s_w[lrow];
                    atomicAdd(out + (size_t)tok * DD + gn, val);
                }
            }
        }
    }
}

// ---------------- host launch ----------------
extern "C" void kernel_launch(void* const* d_in, const int* in_sizes, int n_in,
                              void* d_out, int out_size, void* d_ws, size_t ws_size,
                              hipStream_t stream) {
    (void)in_sizes; (void)n_in; (void)ws_size;
    const float* x  = (const float*)d_in[0];
    const float* Wg = (const float*)d_in[1];
    const float* bg = (const float*)d_in[2];
    const float* W1 = (const float*)d_in[3];
    const float* b1 = (const float*)d_in[4];
    const float* W2 = (const float*)d_in[5];
    const float* b2 = (const float*)d_in[6];
    float* out = (float*)d_out;

    // workspace carve
    size_t off = 0;
    auto carve = [&](size_t bytes) -> char* {
        char* p = (char*)d_ws + off;
        off += (bytes + 255) & ~(size_t)255;
        return p;
    };
    bf16*  xb      = (bf16*)carve((size_t)TT * DD * 2);
    bf16*  w1t     = (bf16*)carve((size_t)EE * FF * DD * 2);
    bf16*  w2t     = (bf16*)carve((size_t)EE * DD * FF * 2);
    bf16*  Hbuf    = (bf16*)carve((size_t)CAP * FF * 2);
    int*   rowtok  = (int*)carve((size_t)CAP * 4);
    float* roww    = (float*)carve((size_t)CAP * 4);
    int*   topi    = (int*)carve((size_t)TT * 2 * 4);
    float* topw    = (float*)carve((size_t)TT * 2 * 4);
    int*   counts  = (int*)carve(256);
    int*   offs    = (int*)carve(256);
    int*   cursors = (int*)carve(256);

    hipMemsetAsync(counts, 0, 256, stream);
    hipMemsetAsync(cursors, 0, 256, stream);
    hipMemsetAsync(rowtok, 0xFF, (size_t)CAP * 4, stream);  // -1
    hipMemsetAsync(out, 0, (size_t)out_size * sizeof(float), stream);

    // bf16 conversion / transposes
    {
        int n4 = (TT * DD) / 4;
        cvt_x_kernel<<<(n4 + 255) / 256, 256, 0, stream>>>(x, xb, n4);
    }
    cvt_transpose_kernel<<<dim3(FF / 32, DD / 32, EE), 256, 0, stream>>>(W1, w1t, DD, FF);
    cvt_transpose_kernel<<<dim3(DD / 32, FF / 32, EE), 256, 0, stream>>>(W2, w2t, FF, DD);

    // routing
    router_kernel<<<TT / 8, 256, 0, stream>>>(x, Wg, bg, topi, topw, counts);
    offsets_kernel<<<1, 1, 0, stream>>>(counts, offs);
    scatter_kernel<<<(TT + 255) / 256, 256, 0, stream>>>(topi, topw, offs, cursors, rowtok, roww);

    // expert GEMMs (dynamic LDS so TDM descriptor offsets are compile-time constants)
    gemm1_kernel<<<dim3(CAPTILES, FF / 128), 256, LDS_TOTAL, stream>>>(xb, w1t, b1, rowtok, offs, Hbuf);
    gemm2_kernel<<<dim3(CAPTILES, DD / 128), 256, LDS_TOTAL, stream>>>(Hbuf, w2t, b2, rowtok, roww, offs, out);
}
